// GATLayerDense_38104949850270
// MI455X (gfx1250) — matrix operand order repaired and checked by reading, not compiled, not run on hardware
//
#include <hip/hip_runtime.h>
#include <hip/hip_bf16.h>

// Problem constants (from reference): B=2, T=16 -> BT=32, N=1024, F_IN=F_OUT=128
#define BT 32
#define NN 1024
#define FF 128
#define NEG_SLOPE 0.2f

typedef __attribute__((ext_vector_type(16))) _Float16 v16h;
typedef __attribute__((ext_vector_type(8)))  _Float16 v8h;
typedef __attribute__((ext_vector_type(8)))  float    v8f;
typedef __attribute__((ext_vector_type(4)))  float    v4f;
typedef __attribute__((ext_vector_type(4)))  int      v4i;

#define GLOBAL_AS __attribute__((address_space(1)))
#define LDS_AS    __attribute__((address_space(3)))

#if defined(__has_builtin)
#if __has_builtin(__builtin_amdgcn_global_load_async_to_lds_b128) && \
    __has_builtin(__builtin_amdgcn_s_wait_asynccnt)
#define HAS_ASYNC_LDS 1
#endif
#endif

// ---------------------------------------------------------------------------
// Kernel 0: W [in][out] fp32 -> Wt [out][in] f16  (B-matrix friendly layout)
// ---------------------------------------------------------------------------
__global__ __launch_bounds__(256) void wprep_kernel(const float* __restrict__ W,
                                                    _Float16* __restrict__ Wt) {
    int idx = blockIdx.x * 256 + threadIdx.x;      // 0 .. 16383
    int i = idx >> 7;                              // input feature (K)
    int o = idx & 127;                             // output feature (N)
    Wt[o * FF + i] = (_Float16)W[i * FF + o];
}

// ---------------------------------------------------------------------------
// Kernel 1: projection h = x @ W via WMMA, scores e_src/e_dst, and store
//           h transposed as f16: ht[bt][f][j] (f-major), coalesced via LDS.
// grid = (BT, N/128), block = 256 (8 waves); each wave: 16 rows x 128 feats
// ---------------------------------------------------------------------------
#define HROW 136   // padded LDS row stride (halfs): 272B, 16B-aligned

__global__ __launch_bounds__(256) void proj_kernel(const float* __restrict__ x,
                                                   const _Float16* __restrict__ Wt,
                                                   const float* __restrict__ a_src,
                                                   const float* __restrict__ a_dst,
                                                   _Float16* __restrict__ ht,
                                                   float* __restrict__ e_src,
                                                   float* __restrict__ e_dst) {
    __shared__ _Float16 hlds[FF * HROW];           // 128 x 136 halfs = 34 KB

    const int bt     = blockIdx.x;
    const int wave   = threadIdx.x >> 5;
    const int lane   = threadIdx.x & 31;
    const int half16 = lane >> 4;          // 0: lanes 0-15, 1: lanes 16-31
    const int lrow   = lane & 15;
    const int jb     = blockIdx.y * 128;               // block's first row
    const int i0     = jb + wave * 16;                 // wave's first row

    // ---- A tiles: x rows (fp32 -> f16). A layout (16-bit, 16x32):
    // lane<16: row=lane, K = {0..7, 16..23}; lane>=16: row=lane-16, K = {8..15, 24..31}
    const int kb = half16 * 8;
    const float* xrow = x + ((size_t)bt * NN + (size_t)(i0 + lrow)) * FF;
    v16h A[4];
#pragma unroll
    for (int ks = 0; ks < 4; ++ks) {
        const float* p = xrow + ks * 32 + kb;
        v4f f0 = *(const v4f*)(p);
        v4f f1 = *(const v4f*)(p + 4);
        v4f f2 = *(const v4f*)(p + 16);
        v4f f3 = *(const v4f*)(p + 20);
        v16h a;
#pragma unroll
        for (int i = 0; i < 4; ++i) {
            a[i]      = (_Float16)f0[i];
            a[4 + i]  = (_Float16)f1[i];
            a[8 + i]  = (_Float16)f2[i];
            a[12 + i] = (_Float16)f3[i];
        }
        A[ks] = a;
    }

    // ---- GEMM: 8 N-tiles x 4 K-steps of v_wmma_f32_16x16x32_f16
    // B layout (16-bit, 32x16): lane<16: n=lane, K={0..7, 16..23}; lane>=16: n=lane-16, K={8..15, 24..31}
    v8f C[8];
#pragma unroll
    for (int nt = 0; nt < 8; ++nt) {
        v8f c;
#pragma unroll
        for (int i = 0; i < 8; ++i) c[i] = 0.0f;
        const _Float16* wp = Wt + (size_t)(nt * 16 + lrow) * FF + half16 * 8;
#pragma unroll
        for (int ks = 0; ks < 4; ++ks) {
            const _Float16* q = wp + ks * 32;
            v8h blo = *(const v8h*)(q);
            v8h bhi = *(const v8h*)(q + 16);
            v16h b;
#pragma unroll
            for (int i = 0; i < 8; ++i) { b[i] = blo[i]; b[8 + i] = bhi[i]; }
            c = __builtin_amdgcn_wmma_f32_16x16x32_f16(false, A[ks], false, b,
                                                       (short)0, c, false, false);
        }
        C[nt] = c;
    }

    // ---- scores: e = h . a   (C layout: VGPR r -> row r / r+8, col = lane&15)
    float asv[8], adv[8];
#pragma unroll
    for (int nt = 0; nt < 8; ++nt) {
        asv[nt] = a_src[nt * 16 + lrow];
        adv[nt] = a_dst[nt * 16 + lrow];
    }
#pragma unroll
    for (int r = 0; r < 8; ++r) {
        float ps = 0.0f, pd = 0.0f;
#pragma unroll
        for (int nt = 0; nt < 8; ++nt) {
            float cc = C[nt][r];
            ps = fmaf(cc, asv[nt], ps);
            pd = fmaf(cc, adv[nt], pd);
        }
#pragma unroll
        for (int off = 1; off < 16; off <<= 1) {
            ps += __shfl_xor(ps, off, 32);
            pd += __shfl_xor(pd, off, 32);
        }
        if (lrow == 0) {
            int row = i0 + r + half16 * 8;
            e_src[bt * NN + row] = ps;
            e_dst[bt * NN + row] = pd;
        }
    }

    // ---- transpose h block through LDS, then coalesced b128 stores
#pragma unroll
    for (int nt = 0; nt < 8; ++nt) {
        int f = nt * 16 + lrow;
#pragma unroll
        for (int r = 0; r < 8; ++r) {
            hlds[f * HROW + wave * 16 + r + half16 * 8] = (_Float16)C[nt][r];
        }
    }
    __syncthreads();

    {
        const int f   = threadIdx.x >> 1;          // 0..127
        const int seg = threadIdx.x & 1;           // 64-half segment
        const _Float16* src = hlds + f * HROW + seg * 64;
        _Float16* dst = ht + (size_t)bt * FF * NN + (size_t)f * NN + jb + seg * 64;
#pragma unroll
        for (int k = 0; k < 8; ++k) {
            *(v8h*)(dst + k * 8) = *(const v8h*)(src + k * 8);
        }
    }
}

// ---------------------------------------------------------------------------
// Kernel 2: gmax[bt] = max_j e_dst[bt][j]   (exact row-max via monotone leaky)
// ---------------------------------------------------------------------------
__global__ __launch_bounds__(256) void gmax_kernel(const float* __restrict__ e_dst,
                                                   float* __restrict__ gmax) {
    __shared__ float red[256];
    int bt = blockIdx.x;
    const float* p = e_dst + bt * NN;
    float m = -3.4e38f;
    for (int i = threadIdx.x; i < NN; i += 256) m = fmaxf(m, p[i]);
    red[threadIdx.x] = m;
    __syncthreads();
    for (int s = 128; s > 0; s >>= 1) {
        if (threadIdx.x < s) red[threadIdx.x] = fmaxf(red[threadIdx.x], red[threadIdx.x + s]);
        __syncthreads();
    }
    if (threadIdx.x == 0) gmax[bt] = red[0];
}

// ---------------------------------------------------------------------------
// Kernel 3: fused flash-style attention: out = softmax(leaky(ei+edj)) @ h
// grid = (BT, N/128), block = 256; each wave: 16 i-rows, full 128 features.
// Per j-step the 8KB h-block is staged ONCE per workgroup into LDS
// (async global->LDS when available), double-buffered; WMMA B-tiles come
// from ds_load_b128 instead of 8x-redundant global loads.
// ---------------------------------------------------------------------------
#define BROW 40    // padded LDS row stride (halfs): 80B, 16B-aligned

__global__ __launch_bounds__(256) void attn_kernel(const _Float16* __restrict__ ht,
                                                   const float* __restrict__ e_src,
                                                   const float* __restrict__ e_dst,
                                                   const float* __restrict__ gmax,
                                                   float* __restrict__ out) {
    __shared__ float    ed_s[NN];                  // 4 KB
    __shared__ _Float16 hbuf[2][FF * BROW];        // 2 x 10 KB double buffer

    const int bt = blockIdx.x;
    for (int i = threadIdx.x; i < NN; i += 256) ed_s[i] = e_dst[bt * NN + i];

    const int wave   = threadIdx.x >> 5;
    const int lane   = threadIdx.x & 31;
    const int half16 = lane >> 4;
    const int lrow   = lane & 15;
    const int i0     = blockIdx.y * 128 + wave * 16;
    const int kb     = half16 * 8;     // A-layout K base for P tile

    const _Float16* hb = ht + (size_t)bt * FF * NN;

    // staging assignment: 512 x 16B chunks per j-step, 2 per thread
    const int c0   = threadIdx.x;                  // chunk ids c0, c0+256
    const int f_a  = c0 >> 2,          p_a = c0 & 3;
    const int f_b  = (c0 + 256) >> 2,  p_b = (c0 + 256) & 3;

#if defined(HAS_ASYNC_LDS)
    // builtin expects (int4 addrspace(1)*, int4 addrspace(3)*, imm, imm)
#define STAGE_ONE(s_, d_)                                                       \
    __builtin_amdgcn_global_load_async_to_lds_b128(                             \
        (GLOBAL_AS v4i*)(s_), (LDS_AS v4i*)(d_), 0, 0)
#define STAGE_WAIT() __builtin_amdgcn_s_wait_asynccnt(0)
#else
#define STAGE_ONE(s_, d_) (*(v8h*)(d_) = *(const v8h*)(s_))
#define STAGE_WAIT() ((void)0)
#endif

    // issue stage of j-step js into hbuf[b]
#define STAGE(bsel, j0_)                                                        \
    do {                                                                        \
        const _Float16* sa = hb + (size_t)f_a * NN + (j0_) + p_a * 8;           \
        const _Float16* sb = hb + (size_t)f_b * NN + (j0_) + p_b * 8;           \
        _Float16* da = &hbuf[bsel][f_a * BROW + p_a * 8];                       \
        _Float16* db = &hbuf[bsel][f_b * BROW + p_b * 8];                       \
        STAGE_ONE(sa, da);                                                      \
        STAGE_ONE(sb, db);                                                      \
    } while (0)

    const float es = e_src[bt * NN + i0 + lrow];   // this lane's row score
    const float gm = gmax[bt];
    float m = es + gm;
    m = (m >= 0.0f) ? m : NEG_SLOPE * m;           // exact row max of leaky(e)

    v8f C[8];
#pragma unroll
    for (int nt = 0; nt < 8; ++nt)
#pragma unroll
        for (int i = 0; i < 8; ++i) C[nt][i] = 0.0f;

    float ssum = 0.0f;

    STAGE(0, 0);                                   // prefetch first block

    for (int js = 0; js < 32; ++js) {              // stream over j in steps of 32
        const int j0  = js * 32;
        const int cur = js & 1;

        STAGE_WAIT();                              // my async LDS writes landed
        __syncthreads();                           // whole block's writes visible
        if (js + 1 < 32) STAGE(1 - cur, j0 + 32);  // overlap next block

        const float* ep = ed_s + j0 + kb;
        v4f e0 = *(const v4f*)(ep);
        v4f e1 = *(const v4f*)(ep + 4);
        v4f e2 = *(const v4f*)(ep + 16);
        v4f e3 = *(const v4f*)(ep + 20);

        v16h aP;
#pragma unroll
        for (int i = 0; i < 4; ++i) {
            float t0 = es + e0[i]; t0 = (t0 >= 0.0f) ? t0 : NEG_SLOPE * t0;
            float t1 = es + e1[i]; t1 = (t1 >= 0.0f) ? t1 : NEG_SLOPE * t1;
            float t2 = es + e2[i]; t2 = (t2 >= 0.0f) ? t2 : NEG_SLOPE * t2;
            float t3 = es + e3[i]; t3 = (t3 >= 0.0f) ? t3 : NEG_SLOPE * t3;
            float p0 = __expf(t0 - m);
            float p1 = __expf(t1 - m);
            float p2 = __expf(t2 - m);
            float p3 = __expf(t3 - m);
            ssum += (p0 + p1) + (p2 + p3);
            aP[i]      = (_Float16)p0;   // K = kb + i
            aP[4 + i]  = (_Float16)p1;   // K = kb + 4 + i
            aP[8 + i]  = (_Float16)p2;   // K = kb + 16 + i
            aP[12 + i] = (_Float16)p3;   // K = kb + 20 + i
        }

#pragma unroll
        for (int nt = 0; nt < 8; ++nt) {
            // B tile from LDS: lane<16: n = nt*16+lane, K(j) = {kb..kb+7, kb+16..kb+23}
            const _Float16* q = &hbuf[cur][(nt * 16 + lrow) * BROW + half16 * 8];
            v8h blo = *(const v8h*)(q);
            v8h bhi = *(const v8h*)(q + 16);
            v16h b;
#pragma unroll
            for (int i = 0; i < 8; ++i) { b[i] = blo[i]; b[8 + i] = bhi[i]; }
            C[nt] = __builtin_amdgcn_wmma_f32_16x16x32_f16(false, aP, false, b,
                                                           (short)0, C[nt], false, false);
        }
    }

    // combine the two K-halves of each row's sum, then normalize and store
    ssum += __shfl_xor(ssum, 16, 32);
    float rinv = 1.0f / ssum;                       // valid for row (lane&15)

    float* ob = out + ((size_t)bt * NN + i0) * FF;
#pragma unroll
    for (int r = 0; r < 8; ++r) {
        int rl = r + half16 * 8;                    // local row of this C VGPR
        float sc = __shfl(rinv, rl, 32);            // ds_bpermute broadcast
#pragma unroll
        for (int nt = 0; nt < 8; ++nt) {
            ob[(size_t)rl * FF + nt * 16 + lrow] = C[nt][r] * sc;
        }
    }
#undef STAGE
#undef STAGE_ONE
#undef STAGE_WAIT
}

// ---------------------------------------------------------------------------
extern "C" void kernel_launch(void* const* d_in, const int* in_sizes, int n_in,
                              void* d_out, int out_size, void* d_ws, size_t ws_size,
                              hipStream_t stream) {
    const float* x     = (const float*)d_in[0];   // [BT, N, F_IN] fp32
    const float* W     = (const float*)d_in[1];   // [F_IN, F_OUT] fp32
    const float* a_src = (const float*)d_in[2];   // [F_OUT]
    const float* a_dst = (const float*)d_in[3];   // [F_OUT]
    float* out = (float*)d_out;                   // [BT, N, F_OUT] fp32

    // workspace layout
    char* ws = (char*)d_ws;
    _Float16* Wt = (_Float16*)ws;                           // 128*128*2   = 32 KB
    _Float16* ht = (_Float16*)(ws + 32768);                 // 32*128*1024*2 = 8 MB
    float* e_src = (float*)(ws + 32768 + (size_t)BT * FF * NN * 2);
    float* e_dst = e_src + BT * NN;
    float* gmax  = e_dst + BT * NN;

    wprep_kernel<<<dim3(64), dim3(256), 0, stream>>>(W, Wt);
    proj_kernel<<<dim3(BT, NN / 128), dim3(256), 0, stream>>>(x, Wt, a_src, a_dst,
                                                              ht, e_src, e_dst);
    gmax_kernel<<<dim3(BT), dim3(256), 0, stream>>>(e_dst, gmax);
    attn_kernel<<<dim3(BT, NN / 128), dim3(256), 0, stream>>>(ht, e_src, e_dst,
                                                              gmax, out);
}
